// LSTM_27934467293363
// MI455X (gfx1250) — compile-verified, matrix operand order checked by reading, not run on hardware
//
#include <hip/hip_runtime.h>

// 2-layer LSTM (B=262144, T=8, D=8, H=16) + FC(16->10) for MI455X (gfx1250).
// One wave32 = one 16-sample batch tile. Gates computed with
// v_wmma_f32_16x16x32_f16: A = [x_t | h_t] (16x32, f16), B = [W_ih;W_hh]^T
// (32x64 f16, 4 N-chunks = i,f,g,o), C/D = f32 accumulators seeded with bias.
// Activations use v_exp_f32 + v_rcp_f32 (fast TRANS path, no IEEE div chain).

typedef _Float16 v16h __attribute__((ext_vector_type(16)));
typedef float    v8f  __attribute__((ext_vector_type(8)));

#define LSTM_T 8
#define NCLS   10
#define WPB    8   // waves per block (256 threads)

__device__ __forceinline__ float frcp_fast(float x) {
    return __builtin_amdgcn_rcpf(x);           // single v_rcp_f32
}
__device__ __forceinline__ float fsigm(float x) {
    return frcp_fast(1.0f + __expf(-x));       // v_exp + v_rcp
}
__device__ __forceinline__ float ftanh(float x) {
    float e = __expf(2.0f * x);
    return 1.0f - 2.0f * frcp_fast(e + 1.0f);  // saturates correctly at +-inf
}

// Wave-internal LDS ordering: DS ops from one wave complete in order; we only
// need the compiler not to reorder and the DS counter drained before reuse.
#define WAVE_SYNC() do {                                   \
    __builtin_amdgcn_wave_barrier();                       \
    asm volatile("s_wait_dscnt 0x0" ::: "memory");         \
    __builtin_amdgcn_wave_barrier();                       \
} while (0)

__global__ __launch_bounds__(256) void lstm_wmma_kernel(
    const float* __restrict__ input,
    const float* __restrict__ Wih0, const float* __restrict__ Whh0,
    const float* __restrict__ bih0, const float* __restrict__ bhh0,
    const float* __restrict__ Wih1, const float* __restrict__ Whh1,
    const float* __restrict__ bih1, const float* __restrict__ bhh1,
    const float* __restrict__ Wfc,  const float* __restrict__ bfc,
    float* __restrict__ out, int ntiles)
{
    __shared__ float xs [WPB][16][LSTM_T * 8]; // staged input, 4KB/wave
    __shared__ float h0T[WPB][16][16];         // layer-0 h, [m][j]
    __shared__ float h1T[WPB][16][16];         // layer-1 h, [m][j]
    __shared__ float wfc_s[NCLS][16];
    __shared__ float bfc_s[NCLS];

    const int tid  = threadIdx.x;
    const int w    = tid >> 5;
    const int lane = tid & 31;
    const int half = lane >> 4;   // K/N half per WMMA 16-bit layout
    const int lm   = lane & 15;

    if (tid < NCLS * 16) wfc_s[tid >> 4][tid & 15] = Wfc[tid];
    if (tid < NCLS)      bfc_s[tid] = bfc[tid];
    __syncthreads();

    // ---- build per-lane WMMA B fragments (weights), resident in VGPRs ----
    // 16-bit matrix per-lane striping: element e holds K = (e<8?e:e+8)+8*half
    v16h B0[4], B1[4];
    float b0s[4], b1s[4];
#pragma unroll
    for (int c = 0; c < 4; ++c) {
        const int n = c * 16 + lm;             // gate-output column 0..63
#pragma unroll
        for (int e = 0; e < 16; ++e) {
            int k = (e < 8 ? e : e + 8) + half * 8;
            // layer 0: K = [x(0..7) | h(8..23) | zero-pad(24..31)]
            int kh = k - 8; kh = kh < 0 ? 0 : (kh > 15 ? 15 : kh); // in-bounds
            float whh0 = Whh0[n * 16 + kh];
            float wih0 = Wih0[n * 8 + (k & 7)];
            float w0 = (k < 8) ? wih0 : ((k < 24) ? whh0 : 0.0f);
            B0[c][e] = (_Float16)w0;
            // layer 1: K = [h_l0(0..15) | h_l1(16..31)] (exactly K=32)
            float w1 = (k < 16) ? Wih1[n * 16 + k] : Whh1[n * 16 + (k - 16)];
            B1[c][e] = (_Float16)w1;
        }
        b0s[c] = bih0[n] + bhh0[n];
        b1s[c] = bih1[n] + bhh1[n];
    }

    const int waveId = blockIdx.x * WPB + w;
    const int nwaves = gridDim.x * WPB;

    for (int tile = waveId; tile < ntiles; tile += nwaves) {
        // ---- stage 16 samples (1024 floats) coalesced into LDS ----
        const float4* src = (const float4*)(input + (size_t)tile * 1024);
        float4* dst = (float4*)&xs[w][0][0];
#pragma unroll
        for (int i = 0; i < 8; ++i) dst[lane + 32 * i] = src[lane + 32 * i];

#pragma unroll
        for (int i = 0; i < 8; ++i) {
            (&h0T[w][0][0])[lane + 32 * i] = 0.0f;
            (&h1T[w][0][0])[lane + 32 * i] = 0.0f;
        }
        float c0s[8], c1s[8];
#pragma unroll
        for (int r = 0; r < 8; ++r) { c0s[r] = 0.0f; c1s[r] = 0.0f; }
        WAVE_SYNC();

        for (int t = 0; t < LSTM_T; ++t) {
            // ================= layer 0 =================
            // A-fragment: row m=lm, element e <-> K=(e<8?e:e+8)+8*half
            v16h a{};
            const float* pLo = half ? &h0T[w][lm][0] : &xs[w][lm][t * 8];
#pragma unroll
            for (int e = 0; e < 8; ++e) a[e] = (_Float16)pLo[e];
#pragma unroll
            for (int e = 8; e < 16; ++e) {
                float hv = h0T[w][lm][e];          // K=16..23 (half 0)
                a[e] = half ? (_Float16)0.0f : (_Float16)hv; // K>=24: pad
            }
            v8f g0, g1, g2, g3;
#pragma unroll
            for (int r = 0; r < 8; ++r) { g0[r]=b0s[0]; g1[r]=b0s[1]; g2[r]=b0s[2]; g3[r]=b0s[3]; }
            g0 = __builtin_amdgcn_wmma_f32_16x16x32_f16(false, a, false, B0[0], (short)0, g0, false, false);
            g1 = __builtin_amdgcn_wmma_f32_16x16x32_f16(false, a, false, B0[1], (short)0, g1, false, false);
            g2 = __builtin_amdgcn_wmma_f32_16x16x32_f16(false, a, false, B0[2], (short)0, g2, false, false);
            g3 = __builtin_amdgcn_wmma_f32_16x16x32_f16(false, a, false, B0[3], (short)0, g3, false, false);

            float hnew[8];
#pragma unroll
            for (int r = 0; r < 8; ++r) {          // sample m = 8*half+r, j=lm
                float iv = fsigm(g0[r]);
                float fv = fsigm(g1[r]);
                float gv = ftanh(g2[r]);
                float ov = fsigm(g3[r]);
                float cc = fv * c0s[r] + iv * gv;
                c0s[r] = cc;
                hnew[r] = ov * ftanh(cc);
            }
            WAVE_SYNC();                           // all A-reads of h0T done
#pragma unroll
            for (int r = 0; r < 8; ++r) h0T[w][8 * half + r][lm] = hnew[r];
            WAVE_SYNC();

            // ================= layer 1 =================
            v16h a1{};
#pragma unroll
            for (int e = 0; e < 8; ++e)
                a1[e] = (_Float16)h0T[w][lm][e + 8 * half];           // K=0..15
#pragma unroll
            for (int e = 8; e < 16; ++e)
                a1[e] = (_Float16)h1T[w][lm][(e - 8) + 8 * half];     // K=16..31
#pragma unroll
            for (int r = 0; r < 8; ++r) { g0[r]=b1s[0]; g1[r]=b1s[1]; g2[r]=b1s[2]; g3[r]=b1s[3]; }
            g0 = __builtin_amdgcn_wmma_f32_16x16x32_f16(false, a1, false, B1[0], (short)0, g0, false, false);
            g1 = __builtin_amdgcn_wmma_f32_16x16x32_f16(false, a1, false, B1[1], (short)0, g1, false, false);
            g2 = __builtin_amdgcn_wmma_f32_16x16x32_f16(false, a1, false, B1[2], (short)0, g2, false, false);
            g3 = __builtin_amdgcn_wmma_f32_16x16x32_f16(false, a1, false, B1[3], (short)0, g3, false, false);
#pragma unroll
            for (int r = 0; r < 8; ++r) {
                float iv = fsigm(g0[r]);
                float fv = fsigm(g1[r]);
                float gv = ftanh(g2[r]);
                float ov = fsigm(g3[r]);
                float cc = fv * c1s[r] + iv * gv;
                c1s[r] = cc;
                hnew[r] = ov * ftanh(cc);
            }
            WAVE_SYNC();
#pragma unroll
            for (int r = 0; r < 8; ++r) h1T[w][8 * half + r][lm] = hnew[r];
            WAVE_SYNC();
        }

        // ---- final FC: out[m][c] = h1_final[m] . Wfc[c] + bfc[c] ----
        // lane = sample lm; each half computes 5 of the 10 classes
        float acc[5];
#pragma unroll
        for (int q = 0; q < 5; ++q) acc[q] = bfc_s[half * 5 + q];
#pragma unroll
        for (int j = 0; j < 16; ++j) {
            float hv = h1T[w][lm][j];
#pragma unroll
            for (int q = 0; q < 5; ++q) acc[q] += hv * wfc_s[half * 5 + q][j];
        }
        float* op = out + ((size_t)tile * 16 + lm) * NCLS + half * 5;
#pragma unroll
        for (int q = 0; q < 5; ++q) op[q] = acc[q];
    }
}

extern "C" void kernel_launch(void* const* d_in, const int* in_sizes, int n_in,
                              void* d_out, int out_size, void* d_ws, size_t ws_size,
                              hipStream_t stream) {
    const float* input = (const float*)d_in[0];
    const float* Wih0  = (const float*)d_in[1];
    const float* Whh0  = (const float*)d_in[2];
    const float* bih0  = (const float*)d_in[3];
    const float* bhh0  = (const float*)d_in[4];
    const float* Wih1  = (const float*)d_in[5];
    const float* Whh1  = (const float*)d_in[6];
    const float* bih1  = (const float*)d_in[7];
    const float* bhh1  = (const float*)d_in[8];
    const float* Wfc   = (const float*)d_in[9];
    const float* bfc   = (const float*)d_in[10];
    float* out = (float*)d_out;

    const int B = in_sizes[0] / 64;   // T*D = 64 floats per sample
    const int ntiles = B / 16;        // 16 samples per wave tile
    int blocks = (ntiles + WPB - 1) / WPB;
    if (blocks > 1024) blocks = 1024; // grid-stride: ~2 tiles/wave at B=262144

    lstm_wmma_kernel<<<blocks, 256, 0, stream>>>(
        input, Wih0, Whh0, bih0, bhh0, Wih1, Whh1, bih1, bhh1,
        Wfc, bfc, out, ntiles);
}
